// MaskedMultiHeadAttention_9560597201130
// MI455X (gfx1250) — matrix operand order repaired and checked by reading, not compiled, not run on hardware
//
#include <hip/hip_runtime.h>
#include <hip/hip_bf16.h>

// Problem constants (from reference)
#define DIM    1024
#define S_CTX  2048
#define NHEADS 16
#define HD     64
#define BATCH  2
#define MROWS  (BATCH * S_CTX)   // 4096 flattened token rows

typedef __attribute__((ext_vector_type(16))) __bf16 v16bf;
typedef __attribute__((ext_vector_type(8)))  __bf16 v8bf;
typedef __attribute__((ext_vector_type(4)))  __bf16 v4bf;
typedef __attribute__((ext_vector_type(2)))  __bf16 v2bf;
typedef __attribute__((ext_vector_type(8)))  float  v8f;
typedef __attribute__((ext_vector_type(8)))  unsigned v8u;

__device__ __forceinline__ v16bf concat8(v8bf lo, v8bf hi) {
  return __builtin_shufflevector(lo, hi, 0,1,2,3,4,5,6,7,8,9,10,11,12,13,14,15);
}

// Pack two f32 into a bf16 pair (v_cvt_pk_bf16_f32).
__device__ __forceinline__ unsigned pack2(float a, float b) {
  v2bf t; t[0] = (__bf16)a; t[1] = (__bf16)b;
  return __builtin_bit_cast(unsigned, t);
}

// ---------------------------------------------------------------------------
// fp32 -> bf16 conversion with optional scale.
// For Wq the scale is log2(e)/sqrt(D): scores are produced directly in the
// log2 domain so softmax needs only raw v_exp_f32 (no per-element mul).
// ---------------------------------------------------------------------------
__global__ void cvt_f32_bf16(const float4* __restrict__ in,
                             v4bf* __restrict__ out, int n4, float scale) {
  int i = blockIdx.x * blockDim.x + threadIdx.x;
  int stride = gridDim.x * blockDim.x;
  for (; i < n4; i += stride) {
    float4 f = in[i];
    v4bf o;
    o[0] = (__bf16)(f.x * scale); o[1] = (__bf16)(f.y * scale);
    o[2] = (__bf16)(f.z * scale); o[3] = (__bf16)(f.w * scale);
    out[i] = o;
  }
}

// ---------------------------------------------------------------------------
// QKV projection: y[m,n] = sum_k X[m,k] * W[n,k] + bscale*bias[n]
// One wave computes a 32x64 patch: 2 M-tiles x 4 N-tiles, 8 WMMAs / K-step.
// VTRANS=0: store [B,H,S,hd] (Q,K).  VTRANS=1: store [B,H,hd,S] (V),
//           lane's 8 rows = 8 consecutive s -> one b128 store per tile.
// grid = (MROWS/256, DIM/64); block = 256 (8 waves)
// ---------------------------------------------------------------------------
template <int VTRANS>
__global__ __launch_bounds__(256)
void qkv_gemm(const __bf16* __restrict__ X, const __bf16* __restrict__ W,
              const float* __restrict__ bias, __bf16* __restrict__ Out,
              float bscale) {
  const int lane = threadIdx.x & 31;
  const int wave = threadIdx.x >> 5;
  const int mb   = blockIdx.x * 256 + wave * 32;   // 2 M-tiles: mb, mb+16
  const int nb   = blockIdx.y * 64;                // 4 N-tiles
  const int half = lane >> 4;
  const int lrow = lane & 15;

  v8f acc[2][4];
#pragma unroll
  for (int m = 0; m < 2; ++m)
#pragma unroll
    for (int t = 0; t < 4; ++t) acc[m][t] = v8f{};

  const __bf16* arow0 = X + (size_t)(mb + lrow) * DIM + (half << 3);
  const __bf16* arow1 = X + (size_t)(mb + 16 + lrow) * DIM + (half << 3);
  const __bf16* brow[4];
#pragma unroll
  for (int t = 0; t < 4; ++t)
    brow[t] = W + (size_t)(nb + t * 16 + lrow) * DIM + (half << 3);

#pragma unroll 2
  for (int kc = 0; kc < DIM; kc += 32) {
    v16bf a0 = concat8(*(const v8bf*)(arow0 + kc), *(const v8bf*)(arow0 + kc + 16));
    v16bf a1 = concat8(*(const v8bf*)(arow1 + kc), *(const v8bf*)(arow1 + kc + 16));
    __builtin_prefetch(arow0 + kc + 256, 0, 1);
#pragma unroll
    for (int t = 0; t < 4; ++t) {
      v16bf b = concat8(*(const v8bf*)(brow[t] + kc), *(const v8bf*)(brow[t] + kc + 16));
      acc[0][t] = __builtin_amdgcn_wmma_f32_16x16x32_bf16(false, a0, false, b,
                                                          (short)0, acc[0][t], false, false);
      acc[1][t] = __builtin_amdgcn_wmma_f32_16x16x32_bf16(false, a1, false, b,
                                                          (short)0, acc[1][t], false, false);
    }
  }

#pragma unroll
  for (int t = 0; t < 4; ++t) {
    const int ncol = nb + t * 16 + lrow;
    const float bn = bias[ncol] * bscale;
    const int h = ncol >> 6;        // head
    const int d = ncol & 63;        // dim within head
#pragma unroll
    for (int m = 0; m < 2; ++m) {
      if (VTRANS) {
        const int r0 = mb + m * 16 + (half << 3);
        const int bi = r0 >> 11;
        const int s0 = r0 & (S_CTX - 1);
        v8bf pk;
#pragma unroll
        for (int j = 0; j < 8; ++j) pk[j] = (__bf16)(acc[m][t][j] + bn);
        *(v8bf*)(Out + (((size_t)bi * NHEADS + h) * HD + d) * S_CTX + s0) = pk;
      } else {
#pragma unroll
        for (int j = 0; j < 8; ++j) {
          const int row = mb + m * 16 + j + (half << 3);
          const int bi  = row >> 11;
          const int s   = row & (S_CTX - 1);
          Out[(((size_t)bi * NHEADS + h) * S_CTX + s) * HD + d] =
              (__bf16)(acc[m][t][j] + bn);
        }
      }
    }
  }
}

// ---------------------------------------------------------------------------
// Flash-attention tile (transposed formulation), 32 keys:
//   S^T = K·Q^T   -> each lane holds 16 scores of ONE q column
//   out^T += V^T·P^T
// Scores arrive pre-multiplied by log2(e)/sqrt(D) (folded into Q), so
// softmax uses raw v_exp_f32 (exp2). Row softmax: in-lane reduction + one
// shfl_xor(16) each for max/sum. No LDS, no fences.
// ---------------------------------------------------------------------------
#define NEGINF (-1e30f)

template <bool MASKED>
__device__ __forceinline__ void attn_tile(
    int kb, int qg, int half, int lrow,
    const __bf16* __restrict__ Kh, const __bf16* __restrict__ Vth,
    v16bf bq0, v16bf bq1,
    float& mrow, float& lsum, v8f (&oaccT)[4]) {
  // --- S^T: D[m=key][n=q].  A = K rows (16 keys x 64d), B = Q^T. 4 WMMAs ---
  v8f t0 = {}, t1 = {};
  {
    const __bf16* ka0 = Kh + (size_t)(kb + lrow) * HD + (half << 3);
    const __bf16* ka1 = Kh + (size_t)(kb + 16 + lrow) * HD + (half << 3);
    v16bf a00 = concat8(*(const v8bf*)(ka0),      *(const v8bf*)(ka0 + 16));
    v16bf a01 = concat8(*(const v8bf*)(ka0 + 32), *(const v8bf*)(ka0 + 48));
    v16bf a10 = concat8(*(const v8bf*)(ka1),      *(const v8bf*)(ka1 + 16));
    v16bf a11 = concat8(*(const v8bf*)(ka1 + 32), *(const v8bf*)(ka1 + 48));
    t0 = __builtin_amdgcn_wmma_f32_16x16x32_bf16(false, a00, false, bq0, (short)0, t0, false, false);
    t0 = __builtin_amdgcn_wmma_f32_16x16x32_bf16(false, a01, false, bq1, (short)0, t0, false, false);
    t1 = __builtin_amdgcn_wmma_f32_16x16x32_bf16(false, a10, false, bq0, (short)0, t1, false, false);
    t1 = __builtin_amdgcn_wmma_f32_16x16x32_bf16(false, a11, false, bq1, (short)0, t1, false, false);
  }

  // --- causal mask (diagonal tiles only): lane's key = kb + j + 8*half (+16)
  if (MASKED) {
#pragma unroll
    for (int j = 0; j < 8; ++j) {
      const int k0 = kb + j + (half << 3);
      if (k0      > qg) t0[j] = NEGINF;
      if (k0 + 16 > qg) t1[j] = NEGINF;
    }
  }

  // --- per-lane max over this lane's 16 scores + one cross-half exchange ---
  float mx = NEGINF;
#pragma unroll
  for (int j = 0; j < 8; ++j) mx = fmaxf(mx, fmaxf(t0[j], t1[j]));
  mx = fmaxf(mx, __shfl_xor(mx, 16, 32));

  const float mnew  = fmaxf(mrow, mx);
  const float alpha = __builtin_amdgcn_exp2f(mrow - mnew);   // log2-domain
  mrow = mnew;

  float p0[8], p1[8];
  float sum = 0.0f;
#pragma unroll
  for (int j = 0; j < 8; ++j) {
    p0[j] = __builtin_amdgcn_exp2f(t0[j] - mnew);
    p1[j] = __builtin_amdgcn_exp2f(t1[j] - mnew);
    sum += p0[j] + p1[j];
  }
  sum += __shfl_xor(sum, 16, 32);
  lsum = lsum * alpha + sum;
#pragma unroll
  for (int nh = 0; nh < 4; ++nh) oaccT[nh] *= alpha;   // per-lane scalar

  // --- build P^T B-operand: pack pairs, one half-exchange, select ---
  unsigned o0[4], o1[4], x0[4], x1[4];
#pragma unroll
  for (int v = 0; v < 4; ++v) {
    o0[v] = pack2(p0[2 * v], p0[2 * v + 1]);
    o1[v] = pack2(p1[2 * v], p1[2 * v + 1]);
  }
#pragma unroll
  for (int v = 0; v < 4; ++v) {
    x0[v] = (unsigned)__shfl_xor((int)o0[v], 16, 32);
    x1[v] = (unsigned)__shfl_xor((int)o1[v], 16, 32);
  }
  v8u bu;
#pragma unroll
  for (int v = 0; v < 4; ++v) {
    bu[v]     = half ? x1[v] : o0[v];   // K 0..7   (keys kb+0..7  / kb+16..23)
    bu[4 + v] = half ? o1[v] : x0[v];   // K 8..15  (keys kb+8..15 / kb+24..31)
  }
  const v16bf bp = __builtin_bit_cast(v16bf, bu);

  // --- out^T += V^T x P^T: A = V^T rows (contiguous from [hd,S] layout) ---
#pragma unroll
  for (int nh = 0; nh < 4; ++nh) {
    const __bf16* vp = Vth + (size_t)(nh * 16 + lrow) * S_CTX + kb + (half << 4);
    v16bf av = *(const v16bf*)(vp);
    oaccT[nh] = __builtin_amdgcn_wmma_f32_16x16x32_bf16(false, av, false, bp,
                                                        (short)0, oaccT[nh], false, false);
  }
}

// ---------------------------------------------------------------------------
// Causal flash attention. Q (pre-scaled by log2(e)/sqrt(D)), K in [B,H,S,hd];
// V transposed [B,H,hd,S]; fp32 output [B,S,D].
// 8 waves/block; each wave owns 16 q-rows (q lives in the lane dimension).
// grid = (S/128, H, B); block = 256.  No shared memory.
// ---------------------------------------------------------------------------
__global__ __launch_bounds__(256)
void attn_fwd(const __bf16* __restrict__ Q, const __bf16* __restrict__ K,
              const __bf16* __restrict__ Vt, float* __restrict__ Out) {
  const int lane = threadIdx.x & 31;
  const int wave = threadIdx.x >> 5;

  const int h  = blockIdx.y;
  const int b  = blockIdx.z;
  const int qb = blockIdx.x * 128 + wave * 16;
  const size_t headOff = ((size_t)(b * NHEADS + h)) * S_CTX * HD;
  const __bf16* Qh  = Q  + headOff;
  const __bf16* Kh  = K  + headOff;
  const __bf16* Vth = Vt + headOff;   // [hd][S] within head

  const int half = lane >> 4;
  const int lrow = lane & 15;
  const int qg   = qb + lrow;         // this lane's q column

  // Q tile as two B-operands (contraction d 0..31 and 32..63)
  const __bf16* qp = Qh + (size_t)qg * HD + (half << 3);
  v16bf bq0 = concat8(*(const v8bf*)(qp), *(const v8bf*)(qp + 16));
  v16bf bq1 = concat8(*(const v8bf*)(qp + 32), *(const v8bf*)(qp + 48));

  float mrow = NEGINF, lsum = 0.0f;
  v8f oaccT[4];
#pragma unroll
  for (int c = 0; c < 4; ++c) oaccT[c] = v8f{};

  // Tiles [0, F) have kb+31 <= qb (min row): no masking needed.
  const int F    = (qb + 1) & ~31;
  const int qmax = qb + 15;

  int kb = 0;
  for (; kb < F; kb += 32)
    attn_tile<false>(kb, qg, half, lrow, Kh, Vth, bq0, bq1, mrow, lsum, oaccT);
  for (; kb <= qmax; kb += 32)
    attn_tile<true>(kb, qg, half, lrow, Kh, Vth, bq0, bq1, mrow, lsum, oaccT);

  // --- normalize, write fp32 out [B,S,H*hd]; 8 consecutive d per chunk ---
  const float inv = 1.0f / lsum;
  float* orow = Out + (((size_t)b * S_CTX + qg) * NHEADS + h) * HD + (half << 3);
#pragma unroll
  for (int nh = 0; nh < 4; ++nh) {
    float4 lo = { oaccT[nh][0] * inv, oaccT[nh][1] * inv,
                  oaccT[nh][2] * inv, oaccT[nh][3] * inv };
    float4 hi = { oaccT[nh][4] * inv, oaccT[nh][5] * inv,
                  oaccT[nh][6] * inv, oaccT[nh][7] * inv };
    *(float4*)(orow + nh * 16)     = lo;
    *(float4*)(orow + nh * 16 + 4) = hi;
  }
}

// ---------------------------------------------------------------------------
// Host launcher
// ---------------------------------------------------------------------------
extern "C" void kernel_launch(void* const* d_in, const int* in_sizes, int n_in,
                              void* d_out, int out_size, void* d_ws, size_t ws_size,
                              hipStream_t stream) {
  const float* x  = (const float*)d_in[0];
  const float* Wq = (const float*)d_in[1];
  const float* bq = (const float*)d_in[2];
  const float* Wk = (const float*)d_in[3];
  const float* bk = (const float*)d_in[4];
  const float* Wv = (const float*)d_in[5];
  const float* bv = (const float*)d_in[6];
  (void)in_sizes; (void)n_in; (void)out_size; (void)ws_size;

  // log2(e)/sqrt(1024): scores computed directly in log2 domain.
  const float scl = 0.03125f * 1.44269504088896340736f;

  char* ws = (char*)d_ws;
  __bf16* xb  = (__bf16*)(ws);                        //  8 MiB: x bf16
  __bf16* wqb = (__bf16*)(ws + (8ull  << 20));        //  2 MiB (pre-scaled)
  __bf16* wkb = (__bf16*)(ws + (10ull << 20));        //  2 MiB
  __bf16* wvb = (__bf16*)(ws + (12ull << 20));        //  2 MiB
  __bf16* Qb  = (__bf16*)(ws + (14ull << 20));        //  8 MiB [B,H,S,hd]
  __bf16* Kb  = (__bf16*)(ws + (22ull << 20));        //  8 MiB [B,H,S,hd]
  __bf16* Vtb = (__bf16*)(ws + (30ull << 20));        //  8 MiB [B,H,hd,S]

  cvt_f32_bf16<<<1024, 256, 0, stream>>>((const float4*)x,  (v4bf*)xb,  MROWS * DIM / 4, 1.0f);
  cvt_f32_bf16<<<256,  256, 0, stream>>>((const float4*)Wq, (v4bf*)wqb, DIM * DIM / 4, scl);
  cvt_f32_bf16<<<256,  256, 0, stream>>>((const float4*)Wk, (v4bf*)wkb, DIM * DIM / 4, 1.0f);
  cvt_f32_bf16<<<256,  256, 0, stream>>>((const float4*)Wv, (v4bf*)wvb, DIM * DIM / 4, 1.0f);

  dim3 ggrid(MROWS / 256, DIM / 64);
  qkv_gemm<0><<<ggrid, 256, 0, stream>>>(xb, wqb, bq, Qb,  scl);
  qkv_gemm<0><<<ggrid, 256, 0, stream>>>(xb, wkb, bk, Kb,  1.0f);
  qkv_gemm<1><<<ggrid, 256, 0, stream>>>(xb, wvb, bv, Vtb, 1.0f);

  dim3 agrid(S_CTX / 128, NHEADS, BATCH);
  attn_fwd<<<agrid, 256, 0, stream>>>(Qb, Kb, Vtb, (float*)d_out);
}
